// GNN_Model_15453292331468
// MI455X (gfx1250) — matrix operand order repaired and checked by reading, not compiled, not run on hardware
//
#include <hip/hip_runtime.h>
#include <hip/hip_bf16.h>

// ---------------------------------------------------------------------------
// Types for CDNA5 WMMA (wave32): v_wmma_f32_16x16x32_bf16
// ---------------------------------------------------------------------------
typedef __attribute__((ext_vector_type(16))) __bf16 v16bf;
typedef __attribute__((ext_vector_type(8)))  float  v8f;

__device__ __forceinline__ unsigned short f2bf(float f) {
    unsigned int u = __float_as_uint(f);
    u = (u + 0x7FFFu + ((u >> 16) & 1u)) >> 16;   // round-to-nearest-even
    return (unsigned short)u;
}
__device__ __forceinline__ unsigned int pk2(float a, float b) {
    return (unsigned int)f2bf(a) | ((unsigned int)f2bf(b) << 16);
}

#define KB  128  // K block per staging step (4 WMMA k-steps of 32)
#define AST 136  // LDS row stride in bf16 halves: 272B = 16B-aligned, 68 banks/row
#define BST 136  //  -> row r starts at bank 4r mod 64: conflict-free for 16 rows

// ---------------------------------------------------------------------------
// Generic GEMM: C[M,N] = A[M,K] @ W[N,K]^T (+bias) (+accum C) (relu) (atomic)
// Block = 256 threads = 8 waves; block tile = 16(M) x 128(N); wave tile 16x16.
// A/B tiles staged in LDS as packed bf16 (converted once at store); fast path
// uses unguarded float4 global loads (16B alignment holds for all call sites).
// grid = (ceil(N/128), ceil(M/16), kSplits); kChunk = K-range per z-slice.
// ---------------------------------------------------------------------------
__global__ __launch_bounds__(256) void gemm_wmma(
    const float* __restrict__ A, const float* __restrict__ W,
    const float* __restrict__ bias, float* __restrict__ C,
    int M, int N, int K, int kChunk, int doRelu, int doAccum, int doAtomic)
{
    __shared__ __align__(16) unsigned short As[16][AST];
    __shared__ __align__(16) unsigned short Bs[128][BST];

    const int tid  = threadIdx.x;
    const int wv   = tid >> 5;
    const int lane = tid & 31;
    const int m0   = blockIdx.y << 4;
    const int n0   = blockIdx.x << 7;
    const int kStart = blockIdx.z * kChunk;
    const int kEnd   = (K < kStart + kChunk) ? K : (kStart + kChunk);

    v8f acc = {};

    const int tr  = tid >> 4;          // 0..15
    const int tc8 = (tid & 15) << 3;   // 0,8,..,120 (8 elems per thread-slot)

    for (int kb = kStart; kb < kEnd; kb += KB) {
        // ---- stage A tile (16 x 128) -> bf16 ----
        {
            const float* ap = A + (size_t)(m0 + tr) * (size_t)K + kb + tc8;
            uint4 q;
            if ((m0 + tr) < M && (kb + tc8 + 7) < K) {   // fast: 2x b128 loads
                float4 a0 = *(const float4*)ap;
                float4 a1 = *(const float4*)(ap + 4);
                q = make_uint4(pk2(a0.x, a0.y), pk2(a0.z, a0.w),
                               pk2(a1.x, a1.y), pk2(a1.z, a1.w));
            } else {
                float v[8];
                const bool mok = (m0 + tr) < M;
                #pragma unroll
                for (int j = 0; j < 8; ++j)
                    v[j] = (mok && (kb + tc8 + j) < K) ? ap[j] : 0.f;
                q = make_uint4(pk2(v[0], v[1]), pk2(v[2], v[3]),
                               pk2(v[4], v[5]), pk2(v[6], v[7]));
            }
            *(uint4*)&As[tr][tc8] = q;
        }
        // ---- stage B tile (128 x 128) from (N,K)-row-major weights -> bf16 ----
        #pragma unroll
        for (int it = 0; it < 8; ++it) {
            int r  = tr + (it << 4);
            int gn = n0 + r;
            const float* wp = W + (size_t)gn * (size_t)K + kb + tc8;
            uint4 q;
            if (gn < N && (kb + tc8 + 7) < K) {          // fast: 2x b128 loads
                float4 b0 = *(const float4*)wp;
                float4 b1 = *(const float4*)(wp + 4);
                q = make_uint4(pk2(b0.x, b0.y), pk2(b0.z, b0.w),
                               pk2(b1.x, b1.y), pk2(b1.z, b1.w));
            } else {
                float v[8];
                #pragma unroll
                for (int j = 0; j < 8; ++j)
                    v[j] = (gn < N && (kb + tc8 + j) < K) ? wp[j] : 0.f;
                q = make_uint4(pk2(v[0], v[1]), pk2(v[2], v[3]),
                               pk2(v[4], v[5]), pk2(v[6], v[7]));
            }
            *(uint4*)&Bs[r][tc8] = q;
        }
        if (kb + KB < kEnd) {   // prefetch next weight tile -> global_prefetch_b8
            int pn = n0 + (tid & 127); if (pn >= N) pn = N - 1;
            __builtin_prefetch(&W[(size_t)pn * (size_t)K + kb + KB], 0, 1);
        }
        __syncthreads();

        // 4 WMMA k-steps; fragments load straight from LDS (no VALU on feed path)
        const int mr   = lane & 15;
        const int asel = (lane & 16) ? 8 : 0;            // A: K halves {0-7,16-23}/{8-15,24-31}
        const int br   = (wv << 4) + (lane & 15);        // B: column N
        const int bsel = (lane & 16) ? 8 : 0;            // B: K halves {0-7,16-23}/{8-15,24-31}
        #pragma unroll
        for (int k0 = 0; k0 < KB; k0 += 32) {
            union { v16bf v; uint4 q[2]; } af, bfm;
            af.q[0]  = *(const uint4*)&As[mr][k0 + asel];
            af.q[1]  = *(const uint4*)&As[mr][k0 + 16 + asel];
            bfm.q[0] = *(const uint4*)&Bs[br][k0 + bsel];
            bfm.q[1] = *(const uint4*)&Bs[br][k0 + 16 + bsel];
            acc = __builtin_amdgcn_wmma_f32_16x16x32_bf16(
                false, af.v, false, bfm.v, (short)0, acc, false, false);
        }
        __syncthreads();
    }

    // epilogue: C layout lane->(M = v + 8*(lane>=16), N = lane&15)
    const int cn  = n0 + (wv << 4) + (lane & 15);
    const int cmb = m0 + ((lane & 16) ? 8 : 0);
    #pragma unroll
    for (int v = 0; v < 8; ++v) {
        int cm = cmb + v;
        if (cm < M && cn < N) {
            size_t idx = (size_t)cm * (size_t)N + cn;
            float val = acc[v];
            if (doAtomic) {
                atomicAdd(&C[idx], val);
            } else {
                if (bias)    val += bias[cn];
                if (doAccum) val += C[idx];
                if (doRelu)  val = fmaxf(val, 0.f);
                C[idx] = val;
            }
        }
    }
}

// ---------------------------------------------------------------------------
// Conv1d (VALID, OC=64) as implicit GEMM with WMMA.
// y[b,oc,h] = relu( sum_{ic,t} x[b,ic,h+t] * w[oc,ic,t] + bias[oc] )
// A = 16 h-positions x K (K = IC*8), B = w as (OC,K) row-major (native layout).
// Block = 128 threads = 4 waves; tile = 16(h) x 64(oc). grid=(hTiles, B).
// ---------------------------------------------------------------------------
__global__ __launch_bounds__(128) void conv1d_wmma(
    const float* __restrict__ x, const float* __restrict__ w,
    const float* __restrict__ bias, float* __restrict__ y,
    int IC, int Hin, int Hout)
{
    __shared__ __align__(16) unsigned short As[16][AST];
    __shared__ __align__(16) unsigned short Bs[64][BST];

    const int tid  = threadIdx.x;
    const int wv   = tid >> 5;
    const int lane = tid & 31;
    const int h0   = blockIdx.x << 4;
    const int b    = blockIdx.y;
    const int K    = IC << 3;   // multiple of 128 for all conv layers here

    v8f acc = {};

    for (int kb = 0; kb < K; kb += KB) {
        // im2col A tile: slot = (row, icq); 8 contiguous taps per slot
        #pragma unroll
        for (int it = 0; it < 2; ++it) {
            int slot = tid + (it << 7);          // 0..255 = 16 rows x 16 ic-quads
            int row = slot >> 4, icq = slot & 15;
            int ic  = (kb >> 3) + icq;
            const float* xp = x + ((size_t)b * IC + ic) * (size_t)Hin + h0 + row;
            float v[8];
            if (h0 + row + 7 < Hin) {            // fast: unguarded contiguous loads
                #pragma unroll
                for (int t = 0; t < 8; ++t) v[t] = xp[t];
            } else {
                #pragma unroll
                for (int t = 0; t < 8; ++t)
                    v[t] = (h0 + row + t < Hin) ? xp[t] : 0.f;
            }
            *(uint4*)&As[row][icq << 3] = make_uint4(pk2(v[0], v[1]), pk2(v[2], v[3]),
                                                     pk2(v[4], v[5]), pk2(v[6], v[7]));
        }
        // weight tile: (OC,IC,8) row-major == (64,K) row-major; always in-bounds
        #pragma unroll
        for (int it = 0; it < 8; ++it) {
            int slot = tid + (it << 7);          // 0..1023 = 64 rows x 16 quads
            int r = slot >> 4, c8 = (slot & 15) << 3;
            const float* wp = w + (size_t)r * (size_t)K + kb + c8;
            float4 b0 = *(const float4*)wp;
            float4 b1 = *(const float4*)(wp + 4);
            *(uint4*)&Bs[r][c8] = make_uint4(pk2(b0.x, b0.y), pk2(b0.z, b0.w),
                                             pk2(b1.x, b1.y), pk2(b1.z, b1.w));
        }
        __syncthreads();

        const int mr   = lane & 15;
        const int asel = (lane & 16) ? 8 : 0;
        const int br   = (wv << 4) + (lane & 15);
        const int bsel = (lane & 16) ? 8 : 0;
        #pragma unroll
        for (int k0 = 0; k0 < KB; k0 += 32) {
            union { v16bf v; uint4 q[2]; } af, bfm;
            af.q[0]  = *(const uint4*)&As[mr][k0 + asel];
            af.q[1]  = *(const uint4*)&As[mr][k0 + 16 + asel];
            bfm.q[0] = *(const uint4*)&Bs[br][k0 + bsel];
            bfm.q[1] = *(const uint4*)&Bs[br][k0 + 16 + bsel];
            acc = __builtin_amdgcn_wmma_f32_16x16x32_bf16(
                false, af.v, false, bfm.v, (short)0, acc, false, false);
        }
        __syncthreads();
    }

    const int oc = (wv << 4) + (lane & 15);
    const int hb = h0 + ((lane & 16) ? 8 : 0);
    #pragma unroll
    for (int v = 0; v < 8; ++v) {
        int h = hb + v;
        if (h < Hout)
            y[((size_t)b * 64 + oc) * (size_t)Hout + h] = fmaxf(acc[v] + bias[oc], 0.f);
    }
}

// ---------------------------------------------------------------------------
// Edge aggregation kernels (segment-sum via fp32 global atomics)
// ---------------------------------------------------------------------------
__global__ void agg3(const float* __restrict__ src, const int* __restrict__ sidx,
                     const int* __restrict__ didx, float* __restrict__ sum,
                     float* __restrict__ cnt, int half)
{
    int e = blockIdx.x * blockDim.x + threadIdx.x;
    if (e >= half) return;
    int s = sidx[2 * e];       // even-parity edges
    int d = didx[2 * e];
    atomicAdd(&sum[(size_t)d * 3 + 0], src[(size_t)s * 3 + 0]);
    atomicAdd(&sum[(size_t)d * 3 + 1], src[(size_t)s * 3 + 1]);
    atomicAdd(&sum[(size_t)d * 3 + 2], src[(size_t)s * 3 + 2]);
    atomicAdd(&cnt[d], 1.0f);
}

__global__ void agg128(const float* __restrict__ src, const int* __restrict__ sidx,
                       const int* __restrict__ didx, int parity,
                       float* __restrict__ sum, float* __restrict__ cnt, int half)
{
    int warp = (blockIdx.x * blockDim.x + threadIdx.x) >> 5;   // one edge per wave32
    int lane = threadIdx.x & 31;
    if (warp >= half) return;
    int s = sidx[2 * warp + parity];
    int d = didx[2 * warp + parity];
    const float* sp = src + (size_t)s * 128;
    float*       dp = sum + (size_t)d * 128;
    #pragma unroll
    for (int f = 0; f < 128; f += 32)
        atomicAdd(&dp[f + lane], sp[f + lane]);
    if (lane == 0) atomicAdd(&cnt[d], 1.0f);
}

__global__ void mean_div(float* __restrict__ sum, const float* __restrict__ cnt,
                         long total, int F)
{
    long i = (long)blockIdx.x * blockDim.x + threadIdx.x;
    if (i >= total) return;
    sum[i] = sum[i] / fmaxf(cnt[i / F], 1.0f);
}

// ---------------------------------------------------------------------------
// SAGE K=3 paths (HD=3: VALU is the right tool, 0.4 GF total)
// ---------------------------------------------------------------------------
__global__ void sage_small(const float* __restrict__ mean3, const float* __restrict__ x3,
                           const float* __restrict__ Wl, const float* __restrict__ bl,
                           const float* __restrict__ Wr, float* __restrict__ out)
{
    long i = blockIdx.x;
    int  n = threadIdx.x;   // 128
    float m0 = mean3[i * 3], m1 = mean3[i * 3 + 1], m2 = mean3[i * 3 + 2];
    float x0 = x3[i * 3],    x1 = x3[i * 3 + 1],    x2 = x3[i * 3 + 2];
    float v = bl[n]
            + m0 * Wl[n * 3] + m1 * Wl[n * 3 + 1] + m2 * Wl[n * 3 + 2]
            + x0 * Wr[n * 3] + x1 * Wr[n * 3 + 1] + x2 * Wr[n * 3 + 2];
    out[i * 128 + n] = fmaxf(v, 0.f);
}

__global__ void add_small_relu(float* __restrict__ y, const float* __restrict__ x3,
                               const float* __restrict__ Wr)
{
    long i = blockIdx.x;
    int  n = threadIdx.x;   // 128
    float v = y[i * 128 + n]
            + x3[i * 3] * Wr[n * 3] + x3[i * 3 + 1] * Wr[n * 3 + 1]
            + x3[i * 3 + 2] * Wr[n * 3 + 2];
    y[i * 128 + n] = fmaxf(v, 0.f);
}

// ---------------------------------------------------------------------------
// LayerNorm over last dim = 128: one wave32 per row, 4 elems/lane
// ---------------------------------------------------------------------------
__global__ __launch_bounds__(256) void layernorm128(
    float* __restrict__ x, const float* __restrict__ g, const float* __restrict__ b,
    int rows)
{
    int row  = (blockIdx.x << 3) + (threadIdx.x >> 5);
    int lane = threadIdx.x & 31;
    if (row >= rows) return;
    float* p = x + (size_t)row * 128 + (lane << 2);
    float4 v = *(const float4*)p;
    float s  = v.x + v.y + v.z + v.w;
    float ss = v.x * v.x + v.y * v.y + v.z * v.z + v.w * v.w;
    #pragma unroll
    for (int off = 16; off > 0; off >>= 1) {
        s  += __shfl_xor(s,  off, 32);
        ss += __shfl_xor(ss, off, 32);
    }
    float mean = s * (1.f / 128.f);
    float var  = ss * (1.f / 128.f) - mean * mean;
    float inv  = rsqrtf(var + 1e-5f);
    int c = lane << 2;
    v.x = (v.x - mean) * inv * g[c + 0] + b[c + 0];
    v.y = (v.y - mean) * inv * g[c + 1] + b[c + 1];
    v.z = (v.z - mean) * inv * g[c + 2] + b[c + 2];
    v.w = (v.w - mean) * inv * g[c + 3] + b[c + 3];
    *(float4*)p = v;
}

__global__ void bias_relu(float* __restrict__ y, const float* __restrict__ bias,
                          int total, int N)
{
    int i = blockIdx.x * blockDim.x + threadIdx.x;
    if (i >= total) return;
    y[i] = fmaxf(y[i] + bias[i % N], 0.f);
}

// Final head: logits = h @ Wd2.T + bd2; softmax. 64 rows.
__global__ void head_softmax(const float* __restrict__ h, const float* __restrict__ W,
                             const float* __restrict__ b, float* __restrict__ out)
{
    int i = threadIdx.x;
    if (i >= 64) return;
    float l0 = b[0], l1 = b[1];
    for (int j = 0; j < 100; ++j) {
        float v = h[i * 100 + j];
        l0 += v * W[j];
        l1 += v * W[100 + j];
    }
    float m  = fmaxf(l0, l1);
    float e0 = __expf(l0 - m), e1 = __expf(l1 - m);
    float s  = e0 + e1;
    out[2 * i + 0] = e0 / s;
    out[2 * i + 1] = e1 / s;
}

// ---------------------------------------------------------------------------
// Host orchestration
// ---------------------------------------------------------------------------
extern "C" void kernel_launch(void* const* d_in, const int* in_sizes, int n_in,
                              void* d_out, int out_size, void* d_ws, size_t ws_size,
                              hipStream_t stream)
{
    (void)n_in; (void)out_size; (void)ws_size;

    const float* x_src  = (const float*)d_in[0];
    const float* x_dst  = (const float*)d_in[1];
    const int*   edge_f = (const int*)  d_in[2];
    const int*   edge_p = (const int*)  d_in[3];
    const float* Wp   = (const float*)d_in[4];
    const float* bp   = (const float*)d_in[5];
    const float* Wf   = (const float*)d_in[6];
    const float* bfv  = (const float*)d_in[7];
    const float* W1l0 = (const float*)d_in[8];
    const float* b1l0 = (const float*)d_in[9];
    const float* W1r0 = (const float*)d_in[10];
    const float* W1l1 = (const float*)d_in[11];
    const float* b1l1 = (const float*)d_in[12];
    const float* W1r1 = (const float*)d_in[13];
    const float* W2l0 = (const float*)d_in[14];
    const float* b2l0 = (const float*)d_in[15];
    const float* W2r0 = (const float*)d_in[16];
    // W2l1/b2l1/W2r1 (17,18,19): SAGE2f result is dead in the reference graph
    const float* ln_g = (const float*)d_in[20];
    const float* ln_b = (const float*)d_in[21];
    const float* cw1  = (const float*)d_in[22];
    const float* cb1  = (const float*)d_in[23];
    const float* cw2  = (const float*)d_in[24];
    const float* cb2  = (const float*)d_in[25];
    const float* cw3  = (const float*)d_in[26];
    const float* cb3  = (const float*)d_in[27];
    const float* Wd1  = (const float*)d_in[28];
    const float* bd1  = (const float*)d_in[29];
    const float* Wd2  = (const float*)d_in[30];
    const float* bd2  = (const float*)d_in[31];

    const int E    = in_sizes[2];
    const int half = E / 2;
    const int NP   = 262144;   // B*PN
    const int NF   = 4096;     // B*FN

    // workspace layout (floats)
    float* w = (float*)d_ws;  size_t o = 0;
    float* xp0    = w + o; o += 786432;         // (64,12288) -> (NP,3)
    float* xf0    = w + o; o += 12288;          // (NF,3)
    float* sum3   = w + o; o += 786432;         // (NP,3) sum->mean
    float* cntp   = w + o; o += 262144;
    float* xp1    = w + o; o += 33554432ULL;    // (NP,128), LN in place
    float* sumf   = w + o; o += 524288;         // (NF,128)
    float* cntf   = w + o; o += 4096;
    float* xf1    = w + o; o += 524288;         // (NF,128), LN in place
    float* sum128 = w + o; o += 33554432ULL;    // (NP,128)
    float* xp2    = w + o; o += 33554432ULL;    // (NP,128) == conv input (64,128,4096)
    float* y1     = w + o; o += 16748544ULL;    // (64,64,4089)
    float* y2     = w + o; o += 16719872ULL;    // (64,64,4082)
    float* y3     = w + o; o += 16691200ULL;    // (64,64,4075)
    float* h1     = w + o; o += 6400;           // (64,100)

    // 1) x_p0 = x_dst @ Wp.T + bp        (M=64, N=12288, K=12288)
    gemm_wmma<<<dim3(96, 4, 1), 256, 0, stream>>>(x_dst, Wp, bp, xp0,
                                                  64, 12288, 12288, 12288, 0, 0, 0);
    // 2) x_f0 = x_src @ Wf.T + bf        (M=64, N=192, K=64)
    gemm_wmma<<<dim3(2, 4, 1), 256, 0, stream>>>(x_src, Wf, bfv, xf0,
                                                 64, 192, 64, 128, 0, 0, 0);

    // 3) SAGE1p aggregation: mean over x_f0 messages into p-nodes (HD=3)
    hipMemsetAsync(sum3, 0, 786432 * sizeof(float), stream);
    hipMemsetAsync(cntp, 0, 262144 * sizeof(float), stream);
    agg3<<<(half + 255) / 256, 256, 0, stream>>>(xf0, edge_f, edge_p, sum3, cntp, half);
    mean_div<<<(786432 + 255) / 256, 256, 0, stream>>>(sum3, cntp, 786432L, 3);
    // 4) x_p1 = relu(mean @ W1l0.T + b1l0 + x_p0 @ W1r0.T)
    sage_small<<<NP, 128, 0, stream>>>(sum3, xp0, W1l0, b1l0, W1r0, xp1);

    // 5) SAGE1f: mean over x_p1 messages into f-nodes (128 feats, odd edges)
    hipMemsetAsync(sumf, 0, 524288 * sizeof(float), stream);
    hipMemsetAsync(cntf, 0, 4096 * sizeof(float), stream);
    agg128<<<(half * 32 + 255) / 256, 256, 0, stream>>>(xp1, edge_p, edge_f, 1,
                                                        sumf, cntf, half);
    mean_div<<<(524288 + 255) / 256, 256, 0, stream>>>(sumf, cntf, 524288L, 128);
    //    x_f1 = mean @ W2l0.T + b2l0 ; then += x_f0 @ W2r0.T ; relu
    gemm_wmma<<<dim3(1, 256, 1), 256, 0, stream>>>(sumf, W2l0, b2l0, xf1,
                                                   NF, 128, 128, 128, 0, 0, 0);
    add_small_relu<<<NF, 128, 0, stream>>>(xf1, xf0, W2r0);

    // 6) LayerNorm (in place)
    layernorm128<<<NP / 8, 256, 0, stream>>>(xp1, ln_g, ln_b, NP);
    layernorm128<<<NF / 8, 256, 0, stream>>>(xf1, ln_g, ln_b, NF);

    // 7) SAGE2p: mean over x_f_ln messages into p-nodes (128 feats, even edges)
    hipMemsetAsync(sum128, 0, 33554432ULL * sizeof(float), stream);
    hipMemsetAsync(cntp, 0, 262144 * sizeof(float), stream);
    agg128<<<(half * 32 + 255) / 256, 256, 0, stream>>>(xf1, edge_f, edge_p, 0,
                                                        sum128, cntp, half);
    mean_div<<<(33554432ULL + 255) / 256, 256, 0, stream>>>(sum128, cntp, 33554432L, 128);
    //    x_p2 = relu(mean @ W1l1.T + b1l1 + x_p_ln @ W1r1.T)
    gemm_wmma<<<dim3(1, 16384, 1), 256, 0, stream>>>(sum128, W1l1, b1l1, xp2,
                                                     NP, 128, 128, 128, 0, 0, 0);
    gemm_wmma<<<dim3(1, 16384, 1), 256, 0, stream>>>(xp1, W1r1, nullptr, xp2,
                                                     NP, 128, 128, 128, 1, 1, 0);
    // (SAGE2f is dead code: x_f unused after this point)

    // 8) conv chain on xp2 reinterpreted as (64,128,4096), implicit-GEMM WMMA
    conv1d_wmma<<<dim3(256, 64, 1), 128, 0, stream>>>(xp2, cw1, cb1, y1, 128, 4096, 4089);
    conv1d_wmma<<<dim3(256, 64, 1), 128, 0, stream>>>(y1,  cw2, cb2, y2,  64, 4089, 4082);
    conv1d_wmma<<<dim3(255, 64, 1), 128, 0, stream>>>(y2,  cw3, cb3, y3,  64, 4082, 4075);

    // 9) dense1: (64,260800)@(100,260800)^T, split-K=32 with atomic reduction
    hipMemsetAsync(h1, 0, 6400 * sizeof(float), stream);
    gemm_wmma<<<dim3(1, 4, 32), 256, 0, stream>>>(y3, Wd1, nullptr, h1,
                                                  64, 100, 260800, 8192, 0, 0, 1);
    bias_relu<<<(6400 + 255) / 256, 256, 0, stream>>>(h1, bd1, 6400, 100);

    // 10) dense2 + softmax -> d_out (64,2)
    head_softmax<<<1, 64, 0, stream>>>(h1, Wd2, bd2, (float*)d_out);
}
